// SegMap_pooling_22823456211403
// MI455X (gfx1250) — matrix operand order repaired and checked by reading, not compiled, not run on hardware
//
#include <hip/hip_runtime.h>
#include <hip/hip_bf16.h>
#include <math.h>

// ---------------------------------------------------------------------------
// CDNA5 (gfx1250) wave32 WMMA implementation of the sparse point-transformer
// stage. All matmuls run on v_wmma_f32_16x16x32_bf16, fed exclusively by
// contiguous 16B vector loads (weights pre-transposed, activations padded).
// ---------------------------------------------------------------------------

typedef __bf16 bf16_t;
typedef __attribute__((ext_vector_type(16))) __bf16 v16bf;
typedef __attribute__((ext_vector_type(8)))  __bf16 v8bf;   // one b128 load
typedef __attribute__((ext_vector_type(8)))  float  v8f;

#define C_DIM 384
#define HEADS 8
#define HD    48
#define OUTC  32
#define NBRK  27
#define KCONV (NBRK * C_DIM)   // 10368

// ---- order-preserving float <-> uint encoding for atomic segment-max ------
__device__ __forceinline__ unsigned enc_f32(float f) {
  unsigned u = __float_as_uint(f);
  return (u & 0x80000000u) ? ~u : (u | 0x80000000u);
}
__device__ __forceinline__ float dec_f32(unsigned u) {
  unsigned v = (u & 0x80000000u) ? (u & 0x7FFFFFFFu) : ~u;
  return __uint_as_float(v);
}

// ---- vector fragment loads -------------------------------------------------
__device__ __forceinline__ v16bf ld16(const bf16_t* __restrict__ p) {
  union { v16bf v; v8bf h[2]; } u;
  u.h[0] = *(const v8bf*)p;
  u.h[1] = *(const v8bf*)(p + 8);
  return u.v;
}

// A (16x32 bf16): lane M = lane&15 ; elem j -> K = (j>=8?16:0)+hi*8+(j&7)
// => two contiguous 8-elem runs at K = hi8 and 16+hi8.
__device__ __forceinline__ v16bf load_a32(const bf16_t* __restrict__ A, long lda,
                                          int m0, int kb) {
  int lane = threadIdx.x & 31;
  int hi8 = ((lane >> 4) & 1) << 3;
  const bf16_t* p = A + (long)(m0 + (lane & 15)) * lda + kb + hi8;
  union { v16bf v; v8bf h[2]; } u;
  u.h[0] = *(const v8bf*)p;
  u.h[1] = *(const v8bf*)(p + 16);
  return u.v;
}

// B (32x16 bf16) from K-contiguous (transposed) storage: lane N = lane&15,
// elem j -> K = (j>=8?16:0)+hi*8+(j&7)  => two 8-elem runs at hi8 and 16+hi8.
__device__ __forceinline__ v16bf load_bT(const bf16_t* __restrict__ BT, long ldk,
                                         int n0, int kb) {
  int lane = threadIdx.x & 31;
  int hi8 = ((lane >> 4) & 1) << 3;
  const bf16_t* p = BT + (long)(n0 + (lane & 15)) * ldk + kb + hi8;
  union { v16bf v; v8bf h[2]; } u;
  u.h[0] = *(const v8bf*)p;
  u.h[1] = *(const v8bf*)(p + 16);
  return u.v;
}

__device__ __forceinline__ v8f wmma_bf16(v16bf a, v16bf b, v8f c) {
  return __builtin_amdgcn_wmma_f32_16x16x32_bf16(false, a, false, b,
                                                 (short)0, c, false, false);
}

// ---------------------------------------------------------------------------
// elementwise f32 -> bf16 (optionally transposed: dst[c*R + r] = src[r*Cc + c])
// ---------------------------------------------------------------------------
__global__ void f32_to_bf16_kernel(const float* __restrict__ src,
                                   bf16_t* __restrict__ dst, long n) {
  long i = (long)blockIdx.x * blockDim.x + threadIdx.x;
  long stride = (long)gridDim.x * blockDim.x;
  for (; i < n; i += stride) dst[i] = (bf16_t)src[i];
}

__global__ void transpose_to_bf16_kernel(const float* __restrict__ src,
                                         bf16_t* __restrict__ dst,
                                         int R, int Cc) {
  long n = (long)R * Cc;
  long i = (long)blockIdx.x * blockDim.x + threadIdx.x;
  long stride = (long)gridDim.x * blockDim.x;
  for (; i < n; i += stride) {
    long r = i / Cc;
    int  c = (int)(i - r * Cc);
    dst[(long)c * R + r] = (bf16_t)src[i];
  }
}

// ---------------------------------------------------------------------------
// init: zero segment-max encodings, seed coord min/max
// ---------------------------------------------------------------------------
__global__ void init_ws_kernel(unsigned* __restrict__ pool_enc, long np,
                               unsigned* __restrict__ emb_enc, long ne,
                               int* __restrict__ bmin, int* __restrict__ bmax,
                               int nb3) {
  long i = (long)blockIdx.x * blockDim.x + threadIdx.x;
  long stride = (long)gridDim.x * blockDim.x;
  if (i < nb3) { bmin[i] = 0x7FFFFFFF; bmax[i] = (int)0x80000000; }
  for (long t = i; t < np + ne; t += stride) {
    if (t < np) pool_enc[t] = 0u;
    else        emb_enc[t - np] = 0u;
  }
}

// ---------------------------------------------------------------------------
// projection GEMM + fused atomic segment-max (pool & emb branches)
// grid = (N/16, C/64, 2), block = 32.  16x64 output per wave (4 WMMA tiles/kb).
// ---------------------------------------------------------------------------
__global__ void proj_segmax_kernel(const bf16_t* __restrict__ X,
                                   const bf16_t* __restrict__ WpT, const float* __restrict__ bp,
                                   const int* __restrict__ invp, unsigned* __restrict__ outp,
                                   const bf16_t* __restrict__ WeT, const float* __restrict__ be,
                                   const int* __restrict__ inve, unsigned* __restrict__ oute) {
  const bf16_t* WT  = blockIdx.z ? WeT : WpT;
  const float* bias = blockIdx.z ? be  : bp;
  const int*   inv  = blockIdx.z ? inve : invp;
  unsigned*    out  = blockIdx.z ? oute : outp;

  int m0 = blockIdx.x * 16, n0 = blockIdx.y * 64;
  int lane = threadIdx.x & 31;
  v8f acc[4] = {{}, {}, {}, {}};
  for (int kb = 0; kb < C_DIM; kb += 32) {
    v16bf a = load_a32(X, C_DIM, m0, kb);
#pragma unroll
    for (int t = 0; t < 4; ++t) {
      v16bf b = load_bT(WT, C_DIM, n0 + t * 16, kb);
      acc[t] = wmma_bf16(a, b, acc[t]);
    }
  }
  int hi = (lane >> 4) & 1, nn = lane & 15;
  int seg[8];
#pragma unroll
  for (int r = 0; r < 8; ++r) seg[r] = inv[m0 + r + hi * 8];
#pragma unroll
  for (int t = 0; t < 4; ++t) {
    int col = n0 + t * 16 + nn;
    float bv = bias[col];
#pragma unroll
    for (int r = 0; r < 8; ++r)
      atomicMax(&out[(long)seg[r] * C_DIM + col], enc_f32(acc[t][r] + bv));
  }
}

// ---------------------------------------------------------------------------
// per-batch coordinate min/max over emb + pool coords
// ---------------------------------------------------------------------------
__global__ void coord_minmax_kernel(const int* __restrict__ emb_coords, int Me,
                                    const int* __restrict__ pool_coords, int Mp,
                                    int* __restrict__ bmin, int* __restrict__ bmax) {
  int i = blockIdx.x * blockDim.x + threadIdx.x;
  if (i >= Me + Mp) return;
  const int* c = (i < Me) ? (emb_coords + 4 * i) : (pool_coords + 4 * (i - Me));
  int b = c[0];
#pragma unroll
  for (int a = 0; a < 3; ++a) {
    atomicMin(&bmin[b * 3 + a], c[1 + a]);
    atomicMax(&bmax[b * 3 + a], c[1 + a]);
  }
}

// ---------------------------------------------------------------------------
// decode segment-max + add sinusoidal PE, emit bf16 (grid = Me+Mp, block 384)
// ---------------------------------------------------------------------------
__global__ void add_pe_kernel(const unsigned* __restrict__ emb_enc,
                              const int* __restrict__ emb_coords,
                              bf16_t* __restrict__ emb_out, int Me,
                              const unsigned* __restrict__ pool_enc,
                              const int* __restrict__ pool_coords,
                              bf16_t* __restrict__ pool_out, int Mp,
                              const int* __restrict__ bmin,
                              const int* __restrict__ bmax) {
  int row = blockIdx.x;
  const unsigned* encb; const int* coords; bf16_t* out; int i;
  if (row < Me) { encb = emb_enc;  coords = emb_coords;  out = emb_out;  i = row; }
  else          { encb = pool_enc; coords = pool_coords; out = pool_out; i = row - Me; }
  int b = coords[4 * i];
  int c = threadIdx.x;              // 0..383
  int a  = c >> 7;
  int rr = c & 127;
  int p  = rr >> 1;
  float x  = (float)coords[4 * i + 1 + a];
  float mn = (float)bmin[b * 3 + a];
  float mx = (float)bmax[b * 3 + a];
  float nrm = (x - mn) / (mx - mn + 1e-6f);
  float t   = __powf(10000.0f, -(float)p * (1.0f / 64.0f));
  float arg = nrm * 6.28318530717958647f * t;
  float pe  = (rr & 1) ? __cosf(arg) : __sinf(arg);
  float val = dec_f32(encb[(long)i * C_DIM + c]) + pe;
  out[(long)i * C_DIM + c] = (bf16_t)val;
}

// ---------------------------------------------------------------------------
// GEMM: out = A(rows x C) @ W(C x C) + bias.  16x64 per wave.
// Writes rows_store (>= rows, mult of 16) rows; pad rows written as zeros so
// downstream consumers can read padded buffers unguarded.
// outF: f32 row-major ; outB: bf16 row-major ; outBT: bf16 col-major (ld = ldT)
// ---------------------------------------------------------------------------
__global__ void gemm_bias_kernel(const bf16_t* __restrict__ A, int rows,
                                 const bf16_t* __restrict__ WT,
                                 const float* __restrict__ bias,
                                 float* __restrict__ outF,
                                 bf16_t* __restrict__ outB,
                                 bf16_t* __restrict__ outBT, int ldT) {
  int m0 = blockIdx.x * 16, n0 = blockIdx.y * 64;
  int lane = threadIdx.x & 31;
  v8f acc[4] = {{}, {}, {}, {}};
  for (int kb = 0; kb < C_DIM; kb += 32) {
    v16bf a = load_a32(A, C_DIM, m0, kb);
#pragma unroll
    for (int t = 0; t < 4; ++t) {
      v16bf b = load_bT(WT, C_DIM, n0 + t * 16, kb);
      acc[t] = wmma_bf16(a, b, acc[t]);
    }
  }
  int hi = (lane >> 4) & 1, nn = lane & 15;
#pragma unroll
  for (int t = 0; t < 4; ++t) {
    int col = n0 + t * 16 + nn;
    float bv = bias[col];
#pragma unroll
    for (int r = 0; r < 8; ++r) {
      int row = m0 + r + hi * 8;
      float v = (row < rows) ? (acc[t][r] + bv) : 0.0f;
      if (outF)  outF[(long)row * C_DIM + col] = v;
      if (outB)  outB[(long)row * C_DIM + col] = (bf16_t)v;
      if (outBT) outBT[(long)col * ldT + row]  = (bf16_t)v;
    }
  }
}

// ---------------------------------------------------------------------------
// flash-style masked cross attention. One wave per (16 query rows, head).
// K/V padded to Me32 rows with zeros; padding masked to -inf before softmax.
// V is consumed pre-transposed (vT[d_global * Me32 + e]) for contiguous loads.
// ---------------------------------------------------------------------------
__global__ void attn_kernel(const bf16_t* __restrict__ qm,
                            const int* __restrict__ pool_coords, int Mp,
                            const bf16_t* __restrict__ km,
                            const bf16_t* __restrict__ vT, int Me32,
                            const int* __restrict__ emb_coords, int Me,
                            bf16_t* __restrict__ ob) {
  __shared__ float sP[16 * 32];
  int h  = blockIdx.y;
  int p0 = blockIdx.x * 16;
  int lane = threadIdx.x & 31;
  int hi = (lane >> 4) & 1, nn = lane & 15;
  int hi8 = hi << 3;
  const float scale = 0.14433756729740645f; // 1/sqrt(48)

  // Q fragments (rows p0..p0+15 zero-padded in q buffer; d padded 48 -> 64)
  v16bf aq0, aq1;
  {
    const bf16_t* qrow = qm + (long)(p0 + nn) * C_DIM + h * HD;
    union { v16bf v; v8bf hh[2]; } u0, u1;
    v8bf z8 = {};
    u0.hh[0] = *(const v8bf*)(qrow + (hi8 << 1));
    u0.hh[1] = *(const v8bf*)(qrow + (hi8 << 1) + 8);
    u1.hh[0] = hi ? z8 : *(const v8bf*)(qrow + 32);   // d = 32..39 (hi: d >= 48 padding)
    u1.hh[1] = hi ? z8 : *(const v8bf*)(qrow + 40);   // d = 40..47 (hi: d >= 48 padding)
    aq0 = u0.v; aq1 = u1.v;
  }

  int pbv[8];
#pragma unroll
  for (int r = 0; r < 8; ++r) {
    int row = p0 + r + hi * 8;
    pbv[r] = (row < Mp) ? pool_coords[4 * row] : -1000000;
  }

  float m_run[8], l_run[8];
  v8f acc0 = {}, acc1 = {}, acc2 = {};
#pragma unroll
  for (int r = 0; r < 8; ++r) { m_run[r] = -__builtin_inff(); l_run[r] = 0.0f; }

  for (int e0 = 0; e0 < Me; e0 += 32) {
    // ---- scores: two 16x16 tiles over key columns e0..e0+31 ----
    v8f S[2];
    int ebatch[2];
#pragma unroll
    for (int t = 0; t < 2; ++t) {
      const bf16_t* krow = km + (long)(e0 + t * 16 + nn) * C_DIM + h * HD;
      v16bf b0 = ld16(krow + (hi8 << 1));       // d = khalf + j, khalf = hi*16
      v16bf b1;
      if (hi) { v16bf z = {}; b1 = z; }          // d = 48..63 padding
      else    { b1 = ld16(krow + 32); }          // d = 32..47
      v8f s = {};
      s = wmma_bf16(aq0, b0, s);
      s = wmma_bf16(aq1, b1, s);
      S[t] = s;
      int ecol = e0 + t * 16 + nn;
      ebatch[t] = (ecol < Me) ? emb_coords[4 * ecol] : 0x7FFFFFFF;
    }
    // ---- mask + scale ----
#pragma unroll
    for (int t = 0; t < 2; ++t) {
      int ecol = e0 + t * 16 + nn;
#pragma unroll
      for (int r = 0; r < 8; ++r) {
        float s = S[t][r] * scale;
        if (ecol >= Me)               s = -__builtin_inff();
        else if (ebatch[t] != pbv[r]) s = -1.0e9f;
        S[t][r] = s;
      }
    }
    // ---- online softmax ----
    float P0[8], P1[8];
#pragma unroll
    for (int r = 0; r < 8; ++r) {
      float rm = fmaxf(S[0][r], S[1][r]);
      rm = fmaxf(rm, __shfl_xor(rm, 1, 32));
      rm = fmaxf(rm, __shfl_xor(rm, 2, 32));
      rm = fmaxf(rm, __shfl_xor(rm, 4, 32));
      rm = fmaxf(rm, __shfl_xor(rm, 8, 32));
      float nm = fmaxf(m_run[r], rm);
      float alpha = __expf(m_run[r] - nm);
      m_run[r] = nm;
      P0[r] = __expf(S[0][r] - nm);
      P1[r] = __expf(S[1][r] - nm);
      float rs = P0[r] + P1[r];
      rs += __shfl_xor(rs, 1, 32);
      rs += __shfl_xor(rs, 2, 32);
      rs += __shfl_xor(rs, 4, 32);
      rs += __shfl_xor(rs, 8, 32);
      l_run[r] = l_run[r] * alpha + rs;
      acc0[r] = acc0[r] * alpha;
      acc1[r] = acc1[r] * alpha;
      acc2[r] = acc2[r] * alpha;
    }
    // ---- C-layout -> A-layout transpose of P via LDS ----
    __syncthreads();
#pragma unroll
    for (int r = 0; r < 8; ++r) {
      sP[(r + hi * 8) * 32 + nn]      = P0[r];
      sP[(r + hi * 8) * 32 + 16 + nn] = P1[r];
    }
    __syncthreads();
    v16bf pa;
#pragma unroll
    for (int j = 0; j < 16; ++j) {
      int k = (hi8 << 1) + j;
      pa[j] = (bf16_t)sP[nn * 32 + k];
    }
    // ---- P @ V : 3 head-dim tiles (vT zero-padded past Me) ----
    const bf16_t* v0p = vT + (long)(h * HD + nn) * Me32 + e0 + (hi8 << 1);
    const bf16_t* v1p = vT + (long)(h * HD + 16 + nn) * Me32 + e0 + (hi8 << 1);
    const bf16_t* v2p = vT + (long)(h * HD + 32 + nn) * Me32 + e0 + (hi8 << 1);
    acc0 = wmma_bf16(pa, ld16(v0p), acc0);
    acc1 = wmma_bf16(pa, ld16(v1p), acc1);
    acc2 = wmma_bf16(pa, ld16(v2p), acc2);
  }
  // ---- normalize + write (zeros into pad rows keeps o buffer clean) ----
#pragma unroll
  for (int r = 0; r < 8; ++r) {
    int row = p0 + r + hi * 8;
    float invl = (row < Mp && l_run[r] > 0.0f) ? 1.0f / l_run[r] : 0.0f;
    long base = (long)row * C_DIM + h * HD;
    ob[base + nn]      = (bf16_t)(acc0[r] * invl);
    ob[base + 16 + nn] = (bf16_t)(acc1[r] * invl);
    ob[base + 32 + nn] = (bf16_t)(acc2[r] * invl);
  }
}

// ---------------------------------------------------------------------------
// residual scatter: new_f = feats + oproj[pool_inv]  (store bf16)
// ---------------------------------------------------------------------------
__global__ void residual_kernel(const float* __restrict__ feats,
                                const float* __restrict__ oproj,
                                const int* __restrict__ pool_inv,
                                bf16_t* __restrict__ newf, long total) {
  long i = (long)blockIdx.x * blockDim.x + threadIdx.x;
  long stride = (long)gridDim.x * blockDim.x;
  for (; i < total; i += stride) {
    long row = i / C_DIM;
    int  c   = (int)(i - row * C_DIM);
    newf[i] = (bf16_t)(feats[i] + oproj[(long)pool_inv[row] * C_DIM + c]);
  }
}

// ---------------------------------------------------------------------------
// gathered 27-tap conv as WMMA GEMM: K = 27*384, OUT = 32.
// Each lane's A fragment is two 8-elem runs; an 8-run never crosses a 384
// boundary, so the neighbor index is constant per run -> 2 b128 gathers.
// ---------------------------------------------------------------------------
__global__ void conv_kernel(const bf16_t* __restrict__ newf,
                            const int* __restrict__ nbr,
                            const bf16_t* __restrict__ WcT,  // [OUT][KCONV]
                            const float* __restrict__ bconv,
                            float* __restrict__ out, int N) {
  int n0 = blockIdx.x * 16;
  int lane = threadIdx.x & 31;
  int hi = (lane >> 4) & 1, nn = lane & 15;
  int hi8 = hi << 3;
  v8f acc0 = {}, acc1 = {};
  const int* nrow = nbr + (long)(n0 + nn) * NBRK;
  for (int kb = 0; kb < KCONV; kb += 32) {
    union { v16bf v; v8bf hh[2]; } ua;
#pragma unroll
    for (int rrun = 0; rrun < 2; ++rrun) {
      int Ks = kb + rrun * 16 + hi8;
      int kn = Ks / C_DIM;
      int c0 = Ks - kn * C_DIM;
      int idx = nrow[kn];
      v8bf z = {};
      ua.hh[rrun] = (idx >= 0) ? *(const v8bf*)(newf + (long)idx * C_DIM + c0) : z;
    }
    v16bf b0 = load_bT(WcT, KCONV, 0, kb);
    v16bf b1 = load_bT(WcT, KCONV, 16, kb);
    acc0 = wmma_bf16(ua.v, b0, acc0);
    acc1 = wmma_bf16(ua.v, b1, acc1);
  }
#pragma unroll
  for (int r = 0; r < 8; ++r) {
    int row = n0 + r + hi * 8;
    out[(long)row * OUTC + nn]      = acc0[r] + bconv[nn];
    out[(long)row * OUTC + 16 + nn] = acc1[r] + bconv[16 + nn];
  }
}

// ---------------------------------------------------------------------------
// host launcher
// ---------------------------------------------------------------------------
extern "C" void kernel_launch(void* const* d_in, const int* in_sizes, int n_in,
                              void* d_out, int out_size, void* d_ws, size_t ws_size,
                              hipStream_t stream) {
  const float* feats   = (const float*)d_in[0];
  const float* W_pool  = (const float*)d_in[1];
  const float* b_pool  = (const float*)d_in[2];
  const float* W_emb   = (const float*)d_in[3];
  const float* b_emb   = (const float*)d_in[4];
  const float* Wq      = (const float*)d_in[5];
  const float* bq      = (const float*)d_in[6];
  const float* Wk      = (const float*)d_in[7];
  const float* bk      = (const float*)d_in[8];
  const float* Wv      = (const float*)d_in[9];
  const float* bv      = (const float*)d_in[10];
  const float* Wo      = (const float*)d_in[11];
  const float* bo      = (const float*)d_in[12];
  const float* W_conv  = (const float*)d_in[13];
  const float* b_conv  = (const float*)d_in[14];
  const int* pool_inv    = (const int*)d_in[15];
  const int* pool_coords = (const int*)d_in[16];
  const int* emb_inv     = (const int*)d_in[17];
  const int* emb_coords  = (const int*)d_in[18];
  const int* nbr_idx     = (const int*)d_in[19];

  const int N    = in_sizes[0] / C_DIM;
  const int Mp   = in_sizes[16] / 4;
  const int Me   = in_sizes[18] / 4;
  const int Mp32 = (Mp + 31) & ~31;
  const int Me32 = (Me + 31) & ~31;

  char* ws = (char*)d_ws;
  size_t off = 0;
  auto alloc = [&](size_t bytes) -> void* {
    void* p = ws + off;
    off += (bytes + 255) & ~(size_t)255;
    return p;
  };

  bf16_t* feats_bf = (bf16_t*)alloc((size_t)N * C_DIM * 2);
  bf16_t* WpT = (bf16_t*)alloc((size_t)C_DIM * C_DIM * 2);
  bf16_t* WeT = (bf16_t*)alloc((size_t)C_DIM * C_DIM * 2);
  bf16_t* WqT = (bf16_t*)alloc((size_t)C_DIM * C_DIM * 2);
  bf16_t* WkT = (bf16_t*)alloc((size_t)C_DIM * C_DIM * 2);
  bf16_t* WvT = (bf16_t*)alloc((size_t)C_DIM * C_DIM * 2);
  bf16_t* WoT = (bf16_t*)alloc((size_t)C_DIM * C_DIM * 2);
  bf16_t* WcT = (bf16_t*)alloc((size_t)OUTC * KCONV * 2);
  unsigned* pool_enc = (unsigned*)alloc((size_t)Mp * C_DIM * 4);
  unsigned* emb_enc  = (unsigned*)alloc((size_t)Me * C_DIM * 4);
  bf16_t* pool_fb = (bf16_t*)alloc((size_t)Mp32 * C_DIM * 2);
  bf16_t* emb_fb  = (bf16_t*)alloc((size_t)Me32 * C_DIM * 2);
  bf16_t* q_bf = (bf16_t*)alloc((size_t)Mp32 * C_DIM * 2);
  bf16_t* k_bf = (bf16_t*)alloc((size_t)Me32 * C_DIM * 2);
  bf16_t* vT   = (bf16_t*)alloc((size_t)C_DIM * Me32 * 2);
  bf16_t* o_bf = (bf16_t*)alloc((size_t)Mp32 * C_DIM * 2);
  float*  oproj   = (float*)alloc((size_t)Mp32 * C_DIM * 4);
  bf16_t* newf_bf = (bf16_t*)alloc((size_t)N * C_DIM * 2);
  int* bmin = (int*)alloc(64 * 3 * 4);
  int* bmax = (int*)alloc(64 * 3 * 4);

  auto blocks_for = [](long n) {
    long b = (n + 255) / 256;
    return (int)(b > 2048 ? 2048 : b);
  };
  // bf16 conversions (weights transposed for K-contiguous B fragments)
  f32_to_bf16_kernel<<<blocks_for((long)N * C_DIM), 256, 0, stream>>>(
      feats, feats_bf, (long)N * C_DIM);
  long wsz = (long)C_DIM * C_DIM;
  transpose_to_bf16_kernel<<<blocks_for(wsz), 256, 0, stream>>>(W_pool, WpT, C_DIM, C_DIM);
  transpose_to_bf16_kernel<<<blocks_for(wsz), 256, 0, stream>>>(W_emb,  WeT, C_DIM, C_DIM);
  transpose_to_bf16_kernel<<<blocks_for(wsz), 256, 0, stream>>>(Wq, WqT, C_DIM, C_DIM);
  transpose_to_bf16_kernel<<<blocks_for(wsz), 256, 0, stream>>>(Wk, WkT, C_DIM, C_DIM);
  transpose_to_bf16_kernel<<<blocks_for(wsz), 256, 0, stream>>>(Wv, WvT, C_DIM, C_DIM);
  transpose_to_bf16_kernel<<<blocks_for(wsz), 256, 0, stream>>>(Wo, WoT, C_DIM, C_DIM);
  transpose_to_bf16_kernel<<<blocks_for((long)KCONV * OUTC), 256, 0, stream>>>(
      W_conv, WcT, KCONV, OUTC);

  init_ws_kernel<<<blocks_for((long)(Mp + Me) * C_DIM), 256, 0, stream>>>(
      pool_enc, (long)Mp * C_DIM, emb_enc, (long)Me * C_DIM, bmin, bmax, 64 * 3);

  // projections + segment max
  {
    dim3 grid(N / 16, C_DIM / 64, 2);
    proj_segmax_kernel<<<grid, 32, 0, stream>>>(feats_bf,
        WpT, b_pool, pool_inv, pool_enc,
        WeT, b_emb,  emb_inv,  emb_enc);
  }

  // per-batch coord min/max, then PE add + decode
  {
    int tot = Me + Mp;
    coord_minmax_kernel<<<(tot + 255) / 256, 256, 0, stream>>>(
        emb_coords, Me, pool_coords, Mp, bmin, bmax);
    add_pe_kernel<<<tot, C_DIM, 0, stream>>>(
        emb_enc, emb_coords, emb_fb, Me,
        pool_enc, pool_coords, pool_fb, Mp, bmin, bmax);
  }

  // QKV projections (pad rows written as zeros)
  {
    dim3 gq(Mp32 / 16, C_DIM / 64);
    dim3 ge(Me32 / 16, C_DIM / 64);
    gemm_bias_kernel<<<gq, 32, 0, stream>>>(pool_fb, Mp, WqT, bq, nullptr, q_bf, nullptr, 0);
    gemm_bias_kernel<<<ge, 32, 0, stream>>>(emb_fb, Me, WkT, bk, nullptr, k_bf, nullptr, 0);
    gemm_bias_kernel<<<ge, 32, 0, stream>>>(emb_fb, Me, WvT, bv, nullptr, nullptr, vT, Me32);
  }

  // attention
  {
    dim3 grid((Mp + 15) / 16, HEADS);
    attn_kernel<<<grid, 32, 0, stream>>>(q_bf, pool_coords, Mp,
                                         k_bf, vT, Me32, emb_coords, Me, o_bf);
  }

  // output projection + residual
  {
    dim3 go(Mp32 / 16, C_DIM / 64);
    gemm_bias_kernel<<<go, 32, 0, stream>>>(o_bf, Mp, WoT, bo, oproj, nullptr, nullptr, 0);
    long tot = (long)N * C_DIM;
    residual_kernel<<<blocks_for(tot), 256, 0, stream>>>(feats, oproj, pool_inv,
                                                         newf_bf, tot);
  }

  // gathered conv -> logits
  conv_kernel<<<N / 16, 32, 0, stream>>>(newf_bf, nbr_idx, WcT, b_conv,
                                         (float*)d_out, N);

  (void)ws_size; (void)n_in; (void)out_size;
}
